// Net_80762565034160
// MI455X (gfx1250) — 8-bit matrix operand order repaired and checked by reading, compiled, not run on hardware
//
#include <hip/hip_runtime.h>
#include <math.h>

typedef int v8i __attribute__((ext_vector_type(8)));

#define B_ROWS 65536
#define IN_F   144
#define HID    1152
#define EPS    1e-5f

#define NT     72
#define KS1    3
#define KS     18
#define MT     4
#define ROWS   64
#define NTPW   9

#define OFF_WF1 0u
#define SZ_WF1  (72u*3u*1024u)
#define OFF_WF2 (OFF_WF1 + SZ_WF1)
#define SZ_WF23 (72u*18u*1024u)
#define OFF_WF3 (OFF_WF2 + SZ_WF23)
#define OFF_C1  (OFF_WF3 + SZ_WF23)
#define OFF_C2  (OFF_C1 + 4608u)
#define OFF_SC3 (OFF_C2 + 4608u)
#define OFF_SH3 (OFF_SC3 + 4608u)
#define OFF_S1  (OFF_SH3 + 4608u)
#define OFF_S2  (OFF_S1 + 1152u)

#define LDS_A     0
#define LDS_B     73728
#define LDS_RS    147456
#define LDS_TOTAL 147968

__global__ void prep_params(const float* b1, const float* g1, const float* be1,
                            const float* m1, const float* v1,
                            const float* b2, const float* g2, const float* be2,
                            const float* m2, const float* v2,
                            const float* b3, const float* g3, const float* be3,
                            const float* m3, const float* v3,
                            char* ws) {
    int n = blockIdx.x * blockDim.x + threadIdx.x;
    if (n >= HID) return;
    float*       c1 = (float*)(ws + OFF_C1);
    float*       c2 = (float*)(ws + OFF_C2);
    float*       sc3 = (float*)(ws + OFF_SC3);
    float*       sh3 = (float*)(ws + OFF_SH3);
    signed char* s1 = (signed char*)(ws + OFF_S1);
    signed char* s2 = (signed char*)(ws + OFF_S2);

    c1[n] = b1[n] - m1[n] + be1[n] * sqrtf(v1[n] + EPS) / g1[n];
    s1[n] = (g1[n] >= 0.f) ? 1 : -1;
    c2[n] = b2[n] - m2[n] + be2[n] * sqrtf(v2[n] + EPS) / g2[n];
    s2[n] = (g2[n] >= 0.f) ? 1 : -1;
    float sc = g3[n] * rsqrtf(v3[n] + EPS);
    sc3[n] = sc;
    sh3[n] = be3[n] + sc * (b3[n] - m3[n]);
}

__global__ void prep_weights(const float* __restrict__ w1,
                             const float* __restrict__ w2,
                             const float* __restrict__ w3,
                             char* __restrict__ ws) {
    unsigned gb = blockIdx.x * 256u + threadIdx.x;
    if (gb >= OFF_C1) return;
    const float* w; unsigned rel; int K; int ksteps; unsigned base;
    if (gb < SZ_WF1)        { w = w1; rel = gb;            K = IN_F; ksteps = KS1; base = OFF_WF1; }
    else if (gb < OFF_WF3)  { w = w2; rel = gb - OFF_WF2;  K = HID;  ksteps = KS;  base = OFF_WF2; }
    else                    { w = w3; rel = gb - OFF_WF3;  K = HID;  ksteps = KS;  base = OFF_WF3; }

    unsigned frag  = rel >> 10;
    unsigned b     = rel & 1023u;
    unsigned lane  = b >> 5;
    unsigned i     = (b >> 2) & 7u;
    unsigned j     = b & 3u;
    unsigned ntile = frag / (unsigned)ksteps;
    unsigned kstep = frag % (unsigned)ksteps;
    int n = (int)(ntile * 16u + (lane & 15u));
    int k = (int)(kstep * 64u + (i >> 1) * 16u + (lane >> 4) * 8u + (i & 1u) * 4u + j);
    signed char v = 0;
    if (k < K) {
        float wv = w[n * K + k];
        v = (wv > 0.f) ? 1 : ((wv < 0.f) ? -1 : 0);
    }
    ws[base + rel] = v;
}

__device__ __forceinline__ v8i load_a_frag(const char* actIn, int stride,
                                           int row, int ks, int kbase8) {
    const char* p = actIn + row * stride + ks * 64 + kbase8;
    unsigned long long q0 = *(const unsigned long long*)(p +  0);
    unsigned long long q1 = *(const unsigned long long*)(p + 16);
    unsigned long long q2 = *(const unsigned long long*)(p + 32);
    unsigned long long q3 = *(const unsigned long long*)(p + 48);
    v8i a;
    a[0] = (int)q0; a[1] = (int)(q0 >> 32);
    a[2] = (int)q1; a[3] = (int)(q1 >> 32);
    a[4] = (int)q2; a[5] = (int)(q2 >> 32);
    a[6] = (int)q3; a[7] = (int)(q3 >> 32);
    return a;
}

__device__ __forceinline__ void bnn_layer(const char* actIn, int strideIn, int ksteps,
                                          const signed char* __restrict__ wfrag,
                                          const float* __restrict__ c,
                                          const signed char* __restrict__ s,
                                          char* actOut, int lane, int wave) {
    const int lrow   = lane & 15;
    const int half   = lane >> 4;
    const int kbase8 = half * 8;
    const v8i vzero  = {0, 0, 0, 0, 0, 0, 0, 0};

    for (int t = 0; t < NTPW; ++t) {
        const int nt = wave * NTPW + t;
        v8i acc[MT];
#pragma unroll
        for (int mt = 0; mt < MT; ++mt) acc[mt] = vzero;

        const v8i* bptr = (const v8i*)(wfrag + (unsigned)nt * ksteps * 1024u);
        for (int ks = 0; ks < ksteps; ++ks) {
            v8i bf = bptr[ks * 32 + lane];
#pragma unroll
            for (int mt = 0; mt < MT; ++mt) {
                v8i af = load_a_frag(actIn, strideIn, mt * 16 + lrow, ks, kbase8);
                acc[mt] = __builtin_amdgcn_wmma_i32_16x16x64_iu8(
                    true, af, true, bf, acc[mt], false, false);
            }
        }
        const int   n  = nt * 16 + lrow;
        const float cn = c[n];
        const float sn = (float)s[n];
#pragma unroll
        for (int mt = 0; mt < MT; ++mt) {
#pragma unroll
            for (int r = 0; r < 8; ++r) {
                const int row = mt * 16 + r + half * 8;
                float yv = ((float)acc[mt][r] + cn) * sn;
                actOut[row * HID + n] = (yv >= 0.f) ? (char)1 : (char)-1;
            }
        }
    }
}

__global__ __launch_bounds__(256) void bnn_fused(
    const float* __restrict__ x, const char* __restrict__ ws,
    const float* __restrict__ w4, const float* __restrict__ b4,
    float* __restrict__ out) {
    extern __shared__ char smem[];
    char*  actA   = smem + LDS_A;
    char*  actB   = smem + LDS_B;
    float* rowsum = (float*)(smem + LDS_RS);

    const int tid  = threadIdx.x;
    const int lane = tid & 31;
    const int wave = tid >> 5;
    const int row0 = blockIdx.x * ROWS;

    for (int idx = tid; idx < ROWS * 192; idx += 256) {
        int r = idx / 192;
        int k = idx - r * 192;
        char v = 0;
        if (k < IN_F) {
            float xv = x[(unsigned)(row0 + r) * IN_F + k];
            v = (xv > 0.f) ? 1 : ((xv < 0.f) ? -1 : 0);
        }
        actA[idx] = v;
    }
    __syncthreads();

    const signed char* wf1 = (const signed char*)(ws + OFF_WF1);
    const signed char* wf2 = (const signed char*)(ws + OFF_WF2);
    const signed char* wf3 = (const signed char*)(ws + OFF_WF3);
    const float*       c1  = (const float*)(ws + OFF_C1);
    const float*       c2  = (const float*)(ws + OFF_C2);
    const float*       sc3 = (const float*)(ws + OFF_SC3);
    const float*       sh3 = (const float*)(ws + OFF_SH3);
    const signed char* s1  = (const signed char*)(ws + OFF_S1);
    const signed char* s2  = (const signed char*)(ws + OFF_S2);

    bnn_layer(actA, 192, KS1, wf1, c1, s1, actB, lane, wave);
    __syncthreads();
    bnn_layer(actB, HID, KS, wf2, c2, s2, actA, lane, wave);
    if (tid < ROWS * 2) rowsum[tid] = 0.f;
    __syncthreads();

    {
        const int lrow   = lane & 15;
        const int half   = lane >> 4;
        const int kbase8 = half * 8;
        const v8i vzero  = {0, 0, 0, 0, 0, 0, 0, 0};
        float P0[MT][8], P1[MT][8];
#pragma unroll
        for (int mt = 0; mt < MT; ++mt)
#pragma unroll
            for (int r = 0; r < 8; ++r) { P0[mt][r] = 0.f; P1[mt][r] = 0.f; }

        for (int t = 0; t < NTPW; ++t) {
            const int nt = wave * NTPW + t;
            v8i acc[MT];
#pragma unroll
            for (int mt = 0; mt < MT; ++mt) acc[mt] = vzero;

            const v8i* bptr = (const v8i*)(wf3 + (unsigned)nt * KS * 1024u);
            for (int ks = 0; ks < KS; ++ks) {
                v8i bf = bptr[ks * 32 + lane];
#pragma unroll
                for (int mt = 0; mt < MT; ++mt) {
                    v8i af = load_a_frag(actA, HID, mt * 16 + lrow, ks, kbase8);
                    acc[mt] = __builtin_amdgcn_wmma_i32_16x16x64_iu8(
                        true, af, true, bf, acc[mt], false, false);
                }
            }
            const int   n   = nt * 16 + lrow;
            const float scn = sc3[n];
            const float shn = sh3[n];
            const float w40 = w4[n];
            const float w41 = w4[HID + n];
#pragma unroll
            for (int mt = 0; mt < MT; ++mt) {
#pragma unroll
                for (int r = 0; r < 8; ++r) {
                    float y = scn * (float)acc[mt][r] + shn;
                    y = fminf(1.f, fmaxf(-1.f, y));
                    P0[mt][r] += y * w40;
                    P1[mt][r] += y * w41;
                }
            }
        }
#pragma unroll
        for (int mt = 0; mt < MT; ++mt) {
#pragma unroll
            for (int r = 0; r < 8; ++r) {
                const int row = mt * 16 + r + half * 8;
                atomicAdd(&rowsum[row * 2 + 0], P0[mt][r]);
                atomicAdd(&rowsum[row * 2 + 1], P1[mt][r]);
            }
        }
    }
    __syncthreads();

    if (tid < ROWS) {
        float l0 = rowsum[tid * 2 + 0] + b4[0];
        float l1 = rowsum[tid * 2 + 1] + b4[1];
        float mx = fmaxf(l0, l1);
        float lse = mx + logf(expf(l0 - mx) + expf(l1 - mx));
        out[(unsigned)(row0 + tid) * 2 + 0] = l0 - lse;
        out[(unsigned)(row0 + tid) * 2 + 1] = l1 - lse;
    }
}

extern "C" void kernel_launch(void* const* d_in, const int* in_sizes, int n_in,
                              void* d_out, int out_size, void* d_ws, size_t ws_size,
                              hipStream_t stream) {
    const float* x   = (const float*)d_in[0];
    const float* w1  = (const float*)d_in[1];
    const float* b1  = (const float*)d_in[2];
    const float* g1  = (const float*)d_in[3];
    const float* be1 = (const float*)d_in[4];
    const float* m1  = (const float*)d_in[5];
    const float* v1  = (const float*)d_in[6];
    const float* w2  = (const float*)d_in[7];
    const float* b2  = (const float*)d_in[8];
    const float* g2  = (const float*)d_in[9];
    const float* be2 = (const float*)d_in[10];
    const float* m2  = (const float*)d_in[11];
    const float* v2  = (const float*)d_in[12];
    const float* w3  = (const float*)d_in[13];
    const float* b3  = (const float*)d_in[14];
    const float* g3  = (const float*)d_in[15];
    const float* be3 = (const float*)d_in[16];
    const float* m3  = (const float*)d_in[17];
    const float* v3  = (const float*)d_in[18];
    const float* w4  = (const float*)d_in[19];
    const float* b4  = (const float*)d_in[20];
    char* ws = (char*)d_ws;

    prep_params<<<dim3((HID + 255) / 256), dim3(256), 0, stream>>>(
        b1, g1, be1, m1, v1, b2, g2, be2, m2, v2, b3, g3, be3, m3, v3, ws);

    prep_weights<<<dim3(OFF_C1 / 256), dim3(256), 0, stream>>>(w1, w2, w3, ws);

    (void)hipFuncSetAttribute((const void*)bnn_fused,
                              hipFuncAttributeMaxDynamicSharedMemorySize,
                              LDS_TOTAL);
    bnn_fused<<<dim3(B_ROWS / ROWS), dim3(256), LDS_TOTAL, stream>>>(
        x, ws, w4, b4, (float*)d_out);
}
